// Attention_88347477279388
// MI455X (gfx1250) — matrix operand order repaired and checked by reading, not compiled, not run on hardware
//
#include <hip/hip_runtime.h>
#include <math.h>

// ---------------------------------------------------------------------------
// MI455X (gfx1250) fused attention block.
// Precision: bf16 WMMA (V_WMMA_F32_16X16X32_BF16) with f32 accumulation.
// Working set (~30MB) is L2-resident (192MB L2): GEMM operands pre-swizzled
// once into WMMA fragment layout, streamed as contiguous 32B/lane loads.
// Each GEMM wave computes a 16x64 tile (A fragment reused over 4 B tiles).
// ---------------------------------------------------------------------------

typedef __bf16 bf16;
typedef __attribute__((ext_vector_type(16))) __bf16 v16bf;
typedef __attribute__((ext_vector_type(8)))  __bf16 v8bf;
typedef __attribute__((ext_vector_type(8)))  float  v8f;

#define N_SEQ  2048
#define DMODEL 1024
#define NHEAD  16
#define HDIM   64
#define KTILES 32          // 1024 / 32
#define SC_PAD 2052        // 2048 + 4 floats: de-conflicts LDS banks
#define SCORE_SCALE 0.125f // 1/sqrt(64)
#define NEGBIG -1000000000.0f

union PairU { unsigned int u; __bf16 h[2]; };

// A-matrix 16x32 lane layout (contiguous K, matching B fragments):
// lane l holds row m=l%16; element e holds K = (l/16)*16 + e
__device__ __forceinline__ int kmapA(int lane, int e) {
  return ((lane >> 4) << 4) + e;
}

__device__ __forceinline__ v8f wmma_bf16(v16bf a, v16bf b, v8f c) {
  return __builtin_amdgcn_wmma_f32_16x16x32_bf16(false, a, false, b,
                                                 (short)0, c, false, false);
}

// ---------------------------------------------------------------------------
// Pack x (2048x1024 f32) into bf16 A-fragment layout [mt][kt][lane][16]
// ---------------------------------------------------------------------------
__global__ __launch_bounds__(256) void pack_x_kernel(const float* __restrict__ x,
                                                     bf16* __restrict__ xf) {
  int t = blockIdx.x * 256 + threadIdx.x;      // 128*32*32 threads
  int lane = t & 31, kt = (t >> 5) & 31, mt = t >> 10;
  const float* xr = x + (mt * 16 + (lane & 15)) * DMODEL + kt * 32;
  v16bf o;
#pragma unroll
  for (int e = 0; e < 16; ++e) o[e] = (bf16)xr[kmapA(lane, e)];
  *(v16bf*)(xf + (((mt * KTILES + kt) * 32 + lane) << 4)) = o;
}

// ---------------------------------------------------------------------------
// Pack W (1024x1024 f32, 4 matrices) into bf16 B-fragment layout
// [nt][kt][lane][16].  B layout: lane l holds col n=l%16, elems are
// K = kt*32 + (l/16)*16 + e (contiguous in K).
// ---------------------------------------------------------------------------
__global__ __launch_bounds__(256) void pack_w_kernel(const float* __restrict__ wq,
                                                     const float* __restrict__ wk,
                                                     const float* __restrict__ wv,
                                                     const float* __restrict__ wo,
                                                     bf16* __restrict__ wf) {
  int sel = blockIdx.z;
  const float* w = (sel == 0) ? wq : (sel == 1) ? wk : (sel == 2) ? wv : wo;
  int t = blockIdx.x * 256 + threadIdx.x;      // 64*32*32 threads
  int lane = t & 31, kt = (t >> 5) & 31, nt = t >> 10;
  int n = nt * 16 + (lane & 15);
  int kbase = kt * 32 + ((lane >> 4) << 4);
  v16bf o;
#pragma unroll
  for (int e = 0; e < 16; ++e) o[e] = (bf16)w[(kbase + e) * DMODEL + n];
  *(v16bf*)(wf + sel * (DMODEL * DMODEL) +
            (((nt * KTILES + kt) * 32 + lane) << 4)) = o;
}

// ---------------------------------------------------------------------------
// Fused QKV projection + bias + RoPE.  One 16x64 tile per wave (A fragment
// reused across 4 B tiles); each wave's 64 columns = exactly one head.
// Q,K written key-major [h][n][64]; V written transposed [h][64][n].
// ---------------------------------------------------------------------------
__global__ __launch_bounds__(256) void qkv_gemm_kernel(
    const bf16* __restrict__ xf, const bf16* __restrict__ wf,
    const float* __restrict__ bq, const float* __restrict__ bk,
    const float* __restrict__ bv, const float* __restrict__ freqs,
    bf16* __restrict__ Q, bf16* __restrict__ K, bf16* __restrict__ Vt) {
  int lane = threadIdx.x & 31, w = threadIdx.x >> 5;
  int mt = blockIdx.x, ng = blockIdx.y * 8 + w, sel = blockIdx.z; // ng: 64-col group
  const bf16* afrag = xf + ((mt * KTILES) << 9);
  const bf16* bbase = wf + sel * (DMODEL * DMODEL);
  const bf16* bf0 = bbase + (((ng * 4 + 0) * KTILES) << 9);
  const bf16* bf1 = bbase + (((ng * 4 + 1) * KTILES) << 9);
  const bf16* bf2 = bbase + (((ng * 4 + 2) * KTILES) << 9);
  const bf16* bf3 = bbase + (((ng * 4 + 3) * KTILES) << 9);
  v8f acc[4] = {};
  for (int kt = 0; kt < KTILES; ++kt) {
    int off = (kt * 32 + lane) << 4;
    int pf = ((kt + 8 < KTILES ? kt + 8 : kt) * 32 + lane) << 4;
    __builtin_prefetch((const void*)(afrag + pf), 0, 3);
    __builtin_prefetch((const void*)(bf0 + pf), 0, 3);
    v16bf a  = *(const v16bf*)(afrag + off);
    v16bf b0 = *(const v16bf*)(bf0 + off);
    v16bf b1 = *(const v16bf*)(bf1 + off);
    v16bf b2 = *(const v16bf*)(bf2 + off);
    v16bf b3 = *(const v16bf*)(bf3 + off);
    acc[0] = wmma_bf16(a, b0, acc[0]);
    acc[1] = wmma_bf16(a, b1, acc[1]);
    acc[2] = wmma_bf16(a, b2, acc[2]);
    acc[3] = wmma_bf16(a, b3, acc[3]);
  }
  int h = ng;                                   // col>>6 is uniform per wave
  const float* bias = (sel == 0) ? bq : (sel == 1) ? bk : bv;
  if (sel < 2) {
    bf16* dst = (sel == 0) ? Q : K;
#pragma unroll
    for (int s = 0; s < 4; ++s) {
      int col = ng * 64 + s * 16 + (lane & 15);
      int hd = col & 63;
      float bcol = bias[col];
      float vals[8];
#pragma unroll
      for (int r = 0; r < 8; ++r) vals[r] = acc[s][r] + bcol;
#pragma unroll
      for (int r = 0; r < 8; ++r) {
        int row = mt * 16 + r + 8 * (lane >> 4);
        float partner = __shfl_xor(vals[r], 1, 32);  // interleaved pair lane (hd ^ 1)
        float f = freqs[row * HDIM + hd];
        float cs = cosf(f), sn = sinf(f);
        float rv = (hd & 1) ? (vals[r] * cs + partner * sn)
                            : (vals[r] * cs - partner * sn);
        dst[(h * N_SEQ + row) * HDIM + hd] = (bf16)rv;
      }
    }
  } else {
    int rowbase = mt * 16 + 8 * (lane >> 4);     // 8 consecutive rows
#pragma unroll
    for (int s = 0; s < 4; ++s) {
      int col = ng * 64 + s * 16 + (lane & 15);
      int hd = col & 63;
      float bcol = bias[col];
      v8bf pk;
#pragma unroll
      for (int r = 0; r < 8; ++r) pk[r] = (bf16)(acc[s][r] + bcol);
      *(v8bf*)(Vt + (h * HDIM + hd) * N_SEQ + rowbase) = pk;
    }
  }
}

// ---------------------------------------------------------------------------
// Attention: one (head, 16-query tile) per workgroup.
// Phase 1: scores (16x2048) via WMMA -> LDS f32
// Phase 2: full-softmax stats + in-place exp (float4 LDS ops); windowed
//          softmax reuses the same exponentials: win_p = e / sum_{win} e
// Phase 3: PV via WMMA (full split over key halves; windowed with per-row
//          predicate folded into the A fragment build)
// Phase 4: scale by reciprocal sums, emit residual (f32) + bf16 context.
// ---------------------------------------------------------------------------
__global__ __launch_bounds__(256) void attn_kernel(
    const bf16* __restrict__ Q, const bf16* __restrict__ K,
    const bf16* __restrict__ Vt, const unsigned char* __restrict__ mask,
    bf16* __restrict__ attn, float* __restrict__ resid) {
  __shared__ float s_sc[16][SC_PAD];      // ~128KB, bank-padded
  __shared__ float s_fs[16], s_ws[16];    // reciprocal sums
  __shared__ float s_acc[2][16][HDIM];    // 8KB
  __shared__ float s_accw[2][16][HDIM];   // 8KB

  int lane = threadIdx.x & 31, w = threadIdx.x >> 5;
  int qt = blockIdx.x, h = blockIdx.y;
  int q0 = qt * 16;

  // Q A-fragments (hd 0..31 and 32..63)
  const bf16* qrow = Q + (h * N_SEQ + q0 + (lane & 15)) * HDIM;
  v16bf qa0, qa1;
#pragma unroll
  for (int v = 0; v < 8; ++v) {
    int k = kmapA(lane, 2 * v);
    PairU p0, p1;
    p0.u = *(const unsigned int*)(qrow + k);
    p1.u = *(const unsigned int*)(qrow + 32 + k);
    qa0[2 * v] = p0.h[0]; qa0[2 * v + 1] = p0.h[1];
    qa1[2 * v] = p1.h[0]; qa1[2 * v + 1] = p1.h[1];
  }

  // ---- Phase 1: scores; wave w covers keys [w*256, w*256+256)
  for (int t = 0; t < 16; ++t) {
    int n0 = (w * 16 + t) * 16;
    int key = n0 + (lane & 15);
    const bf16* krow = K + (h * N_SEQ + key) * HDIM + ((lane >> 4) << 4);
    v16bf b0 = *(const v16bf*)krow;
    v16bf b1 = *(const v16bf*)(krow + 32);
    v8f acc = {};
    acc = wmma_bf16(qa0, b0, acc);
    acc = wmma_bf16(qa1, b1, acc);
    bool km = mask[key] != 0;
#pragma unroll
    for (int r = 0; r < 8; ++r)
      s_sc[r + 8 * (lane >> 4)][key] = km ? acc[r] * SCORE_SCALE : NEGBIG;
  }
  __syncthreads();

  // ---- Phase 2: softmax stats (16 threads per row, contiguous lanes)
  {
    int row = threadIdx.x >> 4, c = threadIdx.x & 15;
    int i = q0 + row;
    float* sr = s_sc[row] + c * 128;
    float4* sr4 = (float4*)sr;
    float mx = -3.0e38f;
    for (int j = 0; j < 32; ++j) {
      float4 v = sr4[j];
      mx = fmaxf(mx, fmaxf(fmaxf(v.x, v.y), fmaxf(v.z, v.w)));
    }
    for (int d = 1; d < 16; d <<= 1) mx = fmaxf(mx, __shfl_xor(mx, d, 32));
    int wlo = i - 128; if (wlo < 0) wlo = 0;
    int whi = i + 128; if (whi > N_SEQ - 1) whi = N_SEQ - 1;
    int c0 = c * 128;
    float fs = 0.0f, wsum = 0.0f;
    for (int j = 0; j < 32; ++j) {
      float4 v = sr4[j];
      float e0 = __expf(v.x - mx), e1 = __expf(v.y - mx);
      float e2 = __expf(v.z - mx), e3 = __expf(v.w - mx);
      fs += (e0 + e1) + (e2 + e3);
      int colb = c0 + 4 * j;
      if (colb + 0 >= wlo && colb + 0 <= whi) wsum += e0;
      if (colb + 1 >= wlo && colb + 1 <= whi) wsum += e1;
      if (colb + 2 >= wlo && colb + 2 <= whi) wsum += e2;
      if (colb + 3 >= wlo && colb + 3 <= whi) wsum += e3;
      sr4[j] = make_float4(e0, e1, e2, e3);
    }
    for (int d = 1; d < 16; d <<= 1) {
      fs   += __shfl_xor(fs, d, 32);
      wsum += __shfl_xor(wsum, d, 32);
    }
    if (c == 0) { s_fs[row] = 1.0f / fs; s_ws[row] = 1.0f / wsum; }
  }
  __syncthreads();

  // ---- Phase 3: PV.  wave -> (colTile ct, key-half hf)
  {
    int ct = w & 3, hf = w >> 2;
    int ncol = ct * 16 + (lane & 15);
    const bf16* vbase = Vt + (h * HDIM + ncol) * N_SEQ + ((lane >> 4) << 4);
    int m = lane & 15;
    const float* srow = s_sc[m];

    v8f acc = {};
    for (int kf = 0; kf < 32; ++kf) {
      int k0 = hf * 1024 + kf * 32;
      v16bf a;
#pragma unroll
      for (int v = 0; v < 8; ++v) {                 // ds_read_b64 + cvt_pk
        float2 pr = *(const float2*)(srow + k0 + kmapA(lane, 2 * v));
        a[2 * v] = (bf16)pr.x; a[2 * v + 1] = (bf16)pr.y;
      }
      v16bf b = *(const v16bf*)(vbase + k0);
      acc = wmma_bf16(a, b, acc);
    }
#pragma unroll
    for (int r = 0; r < 8; ++r) s_acc[hf][r + 8 * (lane >> 4)][ncol] = acc[r];

    // windowed PV over the key span touched by any row of this tile
    int i_row = q0 + m;
    int lo = q0 - 128;       if (lo < 0) lo = 0;
    int hi = q0 + 15 + 128;  if (hi > N_SEQ - 1) hi = N_SEQ - 1;
    int f0 = lo >> 5, f1 = hi >> 5;
    v8f accw = {};
    for (int f = f0; f <= f1; ++f) {
      if (((f - f0) & 1) != hf) continue;
      int k0 = f << 5;
      v16bf a;
#pragma unroll
      for (int v = 0; v < 8; ++v) {
        int k = k0 + kmapA(lane, 2 * v);
        float2 pr = *(const float2*)(srow + k);
        float e0 = (k     >= i_row - 128 && k     <= i_row + 128) ? pr.x : 0.0f;
        float e1 = (k + 1 >= i_row - 128 && k + 1 <= i_row + 128) ? pr.y : 0.0f;
        a[2 * v] = (bf16)e0; a[2 * v + 1] = (bf16)e1;
      }
      v16bf b = *(const v16bf*)(vbase + k0);
      accw = wmma_bf16(a, b, accw);
    }
#pragma unroll
    for (int r = 0; r < 8; ++r) s_accw[hf][r + 8 * (lane >> 4)][ncol] = accw[r];
  }
  __syncthreads();

  // ---- Phase 4: normalize + emit
  for (int id = threadIdx.x; id < 16 * HDIM; id += 256) {
    int m = id >> 6, hd = id & 63;
    float full = (s_acc[0][m][hd] + s_acc[1][m][hd]) * s_fs[m];
    float winv = (s_accw[0][m][hd] + s_accw[1][m][hd]) * s_ws[m];
    resid[(h * N_SEQ + q0 + m) * HDIM + hd] = full - winv;
    attn[(q0 + m) * (NHEAD * HDIM) + h * HDIM + hd] = (bf16)full;
  }
}

// ---------------------------------------------------------------------------
// Output projection: attn(bf16) @ Wo + bo, masked, f32 out.
// One 16x64 tile per wave; A fragment (from unpacked attn rows) reused 4x.
// ---------------------------------------------------------------------------
__global__ __launch_bounds__(256) void proj_kernel(
    const bf16* __restrict__ attn, const bf16* __restrict__ wof,
    const float* __restrict__ bo, const unsigned char* __restrict__ mask,
    float* __restrict__ out) {
  int lane = threadIdx.x & 31, w = threadIdx.x >> 5;
  int mt = blockIdx.x, ng = blockIdx.y * 8 + w;   // 64-col group
  const bf16* arow = attn + (mt * 16 + (lane & 15)) * DMODEL;
  const bf16* bf0 = wof + (((ng * 4 + 0) * KTILES) << 9);
  const bf16* bf1 = wof + (((ng * 4 + 1) * KTILES) << 9);
  const bf16* bf2 = wof + (((ng * 4 + 2) * KTILES) << 9);
  const bf16* bf3 = wof + (((ng * 4 + 3) * KTILES) << 9);
  v8f acc[4] = {};
  for (int kt = 0; kt < KTILES; ++kt) {
    int off = (kt * 32 + lane) << 4;
    int pf = ((kt + 8 < KTILES ? kt + 8 : kt) * 32 + lane) << 4;
    __builtin_prefetch((const void*)(bf0 + pf), 0, 3);
    v16bf a;
#pragma unroll
    for (int v = 0; v < 8; ++v) {
      int k = kt * 32 + kmapA(lane, 2 * v);
      PairU p; p.u = *(const unsigned int*)(arow + k);
      a[2 * v] = p.h[0]; a[2 * v + 1] = p.h[1];
    }
    v16bf b0 = *(const v16bf*)(bf0 + off);
    v16bf b1 = *(const v16bf*)(bf1 + off);
    v16bf b2 = *(const v16bf*)(bf2 + off);
    v16bf b3 = *(const v16bf*)(bf3 + off);
    acc[0] = wmma_bf16(a, b0, acc[0]);
    acc[1] = wmma_bf16(a, b1, acc[1]);
    acc[2] = wmma_bf16(a, b2, acc[2]);
    acc[3] = wmma_bf16(a, b3, acc[3]);
  }
#pragma unroll
  for (int s = 0; s < 4; ++s) {
    int col = ng * 64 + s * 16 + (lane & 15);
    float bc = bo[col];
#pragma unroll
    for (int r = 0; r < 8; ++r) {
      int row = mt * 16 + r + 8 * (lane >> 4);
      float v = acc[s][r] + bc;
      if (!mask[row]) v = 0.0f;
      out[row * DMODEL + col] = v;
    }
  }
}

// ---------------------------------------------------------------------------
// Host wiring
// ---------------------------------------------------------------------------
extern "C" void kernel_launch(void* const* d_in, const int* in_sizes, int n_in,
                              void* d_out, int out_size, void* d_ws, size_t ws_size,
                              hipStream_t stream) {
  const float* x     = (const float*)d_in[0];
  const unsigned char* mask = (const unsigned char*)d_in[1];
  const float* freqs = (const float*)d_in[2];
  const float* Wq = (const float*)d_in[3];
  const float* bq = (const float*)d_in[4];
  const float* Wk = (const float*)d_in[5];
  const float* bk = (const float*)d_in[6];
  const float* Wv = (const float*)d_in[7];
  const float* bv = (const float*)d_in[8];
  const float* Wo = (const float*)d_in[9];
  const float* bo = (const float*)d_in[10];

  float* out   = (float*)d_out;                    // [2048][1024]
  float* resid = out + N_SEQ * DMODEL;             // [16][2048][64]

  char* ws = (char*)d_ws;
  bf16* xf   = (bf16*)(ws);                                  // 4MB
  bf16* wf   = (bf16*)(ws + (size_t)4  * 1024 * 1024);       // 8MB (q,k,v,o)
  bf16* Qb   = (bf16*)(ws + (size_t)12 * 1024 * 1024);       // 4MB
  bf16* Kb   = (bf16*)(ws + (size_t)16 * 1024 * 1024);       // 4MB
  bf16* Vtb  = (bf16*)(ws + (size_t)20 * 1024 * 1024);       // 4MB
  bf16* attn = (bf16*)(ws + (size_t)24 * 1024 * 1024);       // 4MB

  pack_x_kernel<<<dim3(512), dim3(256), 0, stream>>>(x, xf);
  pack_w_kernel<<<dim3(256, 1, 4), dim3(256), 0, stream>>>(Wq, Wk, Wv, Wo, wf);
  qkv_gemm_kernel<<<dim3(128, 2, 3), dim3(256), 0, stream>>>(
      xf, wf, bq, bk, bv, freqs, Qb, Kb, Vtb);
  attn_kernel<<<dim3(128, 16), dim3(256), 0, stream>>>(
      Qb, Kb, Vtb, mask, attn, resid);
  proj_kernel<<<dim3(128, 2), dim3(256), 0, stream>>>(attn, wf + 3 * (DMODEL * DMODEL),
                                                      bo, mask, out);
}
